// RingDilatedAttentionHybrid_18116172054968
// MI455X (gfx1250) — matrix operand order repaired and checked by reading, not compiled, not run on hardware
//
#include <hip/hip_runtime.h>

typedef _Float16 v16h __attribute__((ext_vector_type(16)));
typedef _Float16 h8   __attribute__((ext_vector_type(8)));
typedef float    v8f  __attribute__((ext_vector_type(8)));

#define N_SEQ   8192
#define HEADS   16
#define DHEAD   64
#define HD      (HEADS * DHEAD)      // 1024 floats per sequence position
#define KV_TILE 64
#define QROWS   128                  // rows per workgroup (8 waves x 16)
#define ROW     72                   // padded LDS row (64 + 8) in f16 elements
#define NKT     (N_SEQ / KV_TILE)    // 128 KV tiles

// ---- dilation index math (closed form of the reference gather patterns) ----
__device__ __forceinline__ int q_index(int i, int grp) {
  if (grp == 0) return i;
  if (grp == 1) return (i & ~4095) + 1 + 2 * (i & 2047);
  return 2 + 4 * (i & 2047);
}
__device__ __forceinline__ int k_index(int j, int grp) {
  if (grp == 0) return j;
  if (grp == 1) return (j & ~2047) + 1 + 2 * (j & 1023);
  return (j & ~2047) + 2 + 4 * (j & 511);
}

// reductions across the 16 lanes sharing a row (masks < 16 keep halves separate)
__device__ __forceinline__ float rmax16(float v) {
  v = fmaxf(v, __shfl_xor(v, 1, 32));
  v = fmaxf(v, __shfl_xor(v, 2, 32));
  v = fmaxf(v, __shfl_xor(v, 4, 32));
  v = fmaxf(v, __shfl_xor(v, 8, 32));
  return v;
}
__device__ __forceinline__ float rsum16(float v) {
  v += __shfl_xor(v, 1, 32);
  v += __shfl_xor(v, 2, 32);
  v += __shfl_xor(v, 4, 32);
  v += __shfl_xor(v, 8, 32);
  return v;
}

__device__ __forceinline__ v16h cat8(h8 a, h8 b) {
  return __builtin_shufflevector(a, b, 0,1,2,3,4,5,6,7,8,9,10,11,12,13,14,15);
}

__global__ __launch_bounds__(256)
void ring_dilated_attn(const float* __restrict__ q,
                       const float* __restrict__ k,
                       const float* __restrict__ v,
                       float* __restrict__ out) {
  __shared__ alignas(16) _Float16 Klds[KV_TILE * ROW];   // [key][d]
  __shared__ alignas(16) _Float16 Vlds[DHEAD * ROW];     // [d][key] (transposed)
  __shared__ alignas(16) _Float16 Plds[8 * 16 * ROW];    // per-wave P scratch

  const int head = blockIdx.y;
  const int grp  = (head < 6) ? 0 : (head < 11 ? 1 : 2);
  const int wave = threadIdx.x >> 5;
  const int lane = threadIdx.x & 31;
  const int lm   = lane & 15;
  const int half = lane >> 4;
  const int q_base = blockIdx.x * QROWS + wave * 16;

  // fold 1/sqrt(64) and log2(e) into Q so scores are in exp2 units
  const float qscale = 0.125f * 1.44269504088896f;

  // ---- Q fragments, A-matrix layout (16-bit A: per-lane K packing is
  // [halfbase..+7] then [16+halfbase..+7], halfbase = 8*(lane>=16)) ----
  v16h aq[2];
  {
    const int qrow = q_index(q_base + lm, grp);
    const float* qp = q + (size_t)qrow * HD + head * DHEAD;
    #pragma unroll
    for (int kf = 0; kf < 2; ++kf) {
      v16h a;
      #pragma unroll
      for (int e = 0; e < 8; ++e) {
        a[e]     = (_Float16)(qp[kf * 32 + half * 8 + e]      * qscale);
        a[e + 8] = (_Float16)(qp[kf * 32 + 16 + half * 8 + e] * qscale);
      }
      aq[kf] = a;
    }
  }

  // accumulators in C-layout: vgpr r of tile t holds row (r + 8*half), col t*16+lm
  v8f O[4];
  float m_r[8], l_r[8];
  #pragma unroll
  for (int t = 0; t < 4; ++t) O[t] = (v8f){0.f,0.f,0.f,0.f,0.f,0.f,0.f,0.f};
  #pragma unroll
  for (int r = 0; r < 8; ++r) { m_r[r] = -1e30f; l_r[r] = 0.f; }

  for (int kt = 0; kt < NKT; ++kt) {
    // ---- stage one 64-key tile of K/V into LDS (gathered, f16) ----
    {
      const int kr   = threadIdx.x >> 2;      // 0..63 key row
      const int part = threadIdx.x & 3;       // 4 threads per row
      const int d0   = part * 16;
      const int grow = k_index(kt * KV_TILE + kr, grp);
      const float* kp = k + (size_t)grow * HD + head * DHEAD + d0;
      const float* vp = v + (size_t)grow * HD + head * DHEAD + d0;
      h8 k0, k1;
      #pragma unroll
      for (int e = 0; e < 8; ++e) {
        k0[e] = (_Float16)kp[e];
        k1[e] = (_Float16)kp[8 + e];
      }
      *(h8*)&Klds[kr * ROW + d0]     = k0;
      *(h8*)&Klds[kr * ROW + d0 + 8] = k1;
      #pragma unroll
      for (int e = 0; e < 16; ++e)
        Vlds[(d0 + e) * ROW + kr] = (_Float16)vp[e];
      if (kt + 1 < NKT) {                     // hint next tile into cache
        const int ng = k_index((kt + 1) * KV_TILE + kr, grp);
        __builtin_prefetch(k + (size_t)ng * HD + head * DHEAD + d0, 0, 1);
        __builtin_prefetch(v + (size_t)ng * HD + head * DHEAD + d0, 0, 1);
      }
    }
    __syncthreads();

    // ---- S = Q * K^T  (4 key sub-tiles x K-dim 64 split in two) ----
    v8f S[4];
    #pragma unroll
    for (int t = 0; t < 4; ++t) {
      v8f c = (v8f){0.f,0.f,0.f,0.f,0.f,0.f,0.f,0.f};
      #pragma unroll
      for (int kf = 0; kf < 2; ++kf) {
        // B layout: lane n = lm, K [8*half..+7] then [16+8*half..+7]
        const _Float16* bp = &Klds[(t * 16 + lm) * ROW + kf * 32];
        v16h b = cat8(*(const h8*)(bp + half * 8), *(const h8*)(bp + 16 + half * 8));
        c = __builtin_amdgcn_wmma_f32_16x16x32_f16(false, aq[kf], false, b,
                                                   (short)0, c, false, false);
      }
      S[t] = c;
    }

    // ---- online softmax (rows live across 16 lanes of each half-wave) ----
    float corr[8];
    #pragma unroll
    for (int r = 0; r < 8; ++r) {
      float mx = fmaxf(fmaxf(S[0][r], S[1][r]), fmaxf(S[2][r], S[3][r]));
      mx = rmax16(mx);
      const float mnew = fmaxf(m_r[r], mx);
      const float cf = __builtin_amdgcn_exp2f(m_r[r] - mnew);
      float rs = 0.f;
      #pragma unroll
      for (int t = 0; t < 4; ++t) {
        const float p = __builtin_amdgcn_exp2f(S[t][r] - mnew);
        S[t][r] = p;
        rs += p;
      }
      rs = rsum16(rs);
      l_r[r] = l_r[r] * cf + rs;
      m_r[r] = mnew;
      corr[r] = cf;
    }
    #pragma unroll
    for (int t = 0; t < 4; ++t)
      #pragma unroll
      for (int r = 0; r < 8; ++r) O[t][r] *= corr[r];

    // ---- convert P from C-layout to A-layout via per-wave LDS scratch ----
    _Float16* P = &Plds[wave * 16 * ROW];
    #pragma unroll
    for (int t = 0; t < 4; ++t)
      #pragma unroll
      for (int r = 0; r < 8; ++r)
        P[(r + half * 8) * ROW + t * 16 + lm] = (_Float16)S[t][r];

    // ---- O += P * V ----
    #pragma unroll
    for (int kf = 0; kf < 2; ++kf) {
      const _Float16* pp = &P[lm * ROW + kf * 32];
      v16h ap = cat8(*(const h8*)(pp + half * 8),
                     *(const h8*)(pp + 16 + half * 8));
      #pragma unroll
      for (int nb = 0; nb < 4; ++nb) {
        // B layout: lane n = d-col = nb*16+lm; keys [8*half..+7] then [16+8*half..+7]
        const _Float16* bp = &Vlds[(nb * 16 + lm) * ROW + kf * 32];
        v16h b = cat8(*(const h8*)(bp + half * 8), *(const h8*)(bp + 16 + half * 8));
        O[nb] = __builtin_amdgcn_wmma_f32_16x16x32_f16(false, ap, false, b,
                                                       (short)0, O[nb], false, false);
      }
    }
    __syncthreads();
  }

  // ---- finalize: divide by l, scatter C-layout to (b, n, h, d) ----
  #pragma unroll
  for (int t = 0; t < 4; ++t) {
    #pragma unroll
    for (int r = 0; r < 8; ++r) {
      const int row = q_base + r + half * 8;
      out[(size_t)row * HD + head * DHEAD + t * 16 + lm] = O[t][r] / l_r[r];
    }
  }
}

extern "C" void kernel_launch(void* const* d_in, const int* in_sizes, int n_in,
                              void* d_out, int out_size, void* d_ws, size_t ws_size,
                              hipStream_t stream) {
  const float* q = (const float*)d_in[0];
  const float* k = (const float*)d_in[1];
  const float* v = (const float*)d_in[2];
  float* out = (float*)d_out;
  dim3 grid(N_SEQ / QROWS, HEADS);
  ring_dilated_attn<<<grid, 256, 0, stream>>>(q, k, v, out);
}
